// MultiFeatureGATFusion_30571577213151
// MI455X (gfx1250) — matrix operand order repaired and checked by reading, not compiled, not run on hardware
//
#include <hip/hip_runtime.h>

typedef __attribute__((ext_vector_type(16))) __bf16 v16bf;
typedef __attribute__((ext_vector_type(8)))  __bf16 v8bf;
typedef __attribute__((ext_vector_type(8)))  float  v8f;
typedef __attribute__((ext_vector_type(4)))  float  v4f;

#define DFEAT 256
#define HCNT  4
#define MT    64            // rows per workgroup (16 groups of 4)
#define NWAVE 16            // 512 threads
#define NEG_SLOPE 0.2f

// ---- one-time: W [L=2][256][1024] f32 -> Wt bf16 column-major [l][col][k] ----
__global__ __launch_bounds__(256) void prep_w_kernel(const float* __restrict__ W,
                                                     __bf16* __restrict__ Wt) {
  int idx = blockIdx.x * 256 + threadIdx.x;          // 0 .. 2*256*1024-1
  int l   = idx >> 18;                               // / (256*1024)
  int rem = idx & ((1 << 18) - 1);
  int k   = rem >> 10;
  int col = rem & 1023;
  Wt[(((size_t)l * 1024 + col) << 8) + k] = (__bf16)W[idx];
}

// ---- pack specific[3][B][256] + shared[1][B][256] -> x0[N][256], node r = b*4+n ----
__global__ __launch_bounds__(256) void pack_kernel(const float* __restrict__ spec,
                                                   const float* __restrict__ shar,
                                                   float* __restrict__ x0, int B) {
  int id = blockIdx.x * 256 + threadIdx.x;           // over N*64 float4 units
  int r  = id >> 6;
  int dq = id & 63;
  int b = r >> 2, n = r & 3;
  const float* src = (n < 3) ? (spec + (((size_t)n * B + b) << 8))
                             : (shar + ((size_t)b << 8));
  v4f v = *(const v4f*)(src + dq * 4);
  *(v4f*)(x0 + (((size_t)r) << 8) + dq * 4) = v;
}

// ---- fused GAT layer: GEMM (bf16 WMMA) + group-of-4 softmax attention + head mean + bias + ELU ----
// 512 threads = 16 waves; wave w owns d-tile w (16 cols), all 4 heads, 4 M-tiles (64 rows).
// Each B fragment is reused by 4 WMMAs (4 M-tiles) -> halves L2 traffic on W vs MT=32.
__global__ __launch_bounds__(512) void gat_layer_kernel(
    const float* __restrict__ x_in, const __bf16* __restrict__ Wt,
    const float* __restrict__ attS, const float* __restrict__ attD,
    const float* __restrict__ bias, float* __restrict__ x_out) {
  __shared__ __align__(16) __bf16 sA[MT][DFEAT + 8];        // A tile, bf16 (~33 KB)
  __shared__ __align__(16) float  sAS[MT][HCNT];            // a_src per row/head
  __shared__ __align__(16) float  sAD[MT][HCNT];            // a_dst per row/head
  __shared__ __align__(16) float  sAlpha[16][HCNT][4][4];   // [group][head][j(dst)][i(src)]

  const int tid    = threadIdx.x;
  const int lane   = tid & 31;
  const int wave   = tid >> 5;                 // 0..15 == d-tile index
  const int laneN  = lane & 15;
  const int laneHi = lane >> 4;
  const size_t rowBase = (size_t)blockIdx.x * MT;
  const int dcol = wave * 16 + laneN;          // this lane's output column (0..255)

  if (tid < MT * HCNT) { ((float*)sAS)[tid] = 0.f; ((float*)sAD)[tid] = 0.f; }

  // stage x tile -> bf16 LDS (64 rows x 256 cols, 32 floats/thread)
  {
    int r  = tid >> 3;                         // 0..63
    int c0 = (tid & 7) * 32;                   // 0..224
    const float* src = x_in + (rowBase + r) * DFEAT + c0;
#pragma unroll
    for (int q = 0; q < 8; ++q) {
      v4f v = *(const v4f*)(src + q * 4);
      __bf16* dst = &sA[r][c0 + q * 4];
      dst[0] = (__bf16)v.x; dst[1] = (__bf16)v.y;
      dst[2] = (__bf16)v.z; dst[3] = (__bf16)v.w;
    }
  }
  __syncthreads();

  v8f acc[4][HCNT];                            // [m-tile][head], 128 VGPRs
  v8f vzero = {0.f, 0.f, 0.f, 0.f, 0.f, 0.f, 0.f, 0.f};
#pragma unroll
  for (int m = 0; m < 4; ++m)
#pragma unroll
    for (int h = 0; h < HCNT; ++h) acc[m][h] = vzero;

#pragma unroll 1
  for (int ks = 0; ks < DFEAT; ks += 32) {
    // A fragments (16x32 bf16 layout: lanes 0-15 K={0..7,16..23}, lanes 16-31 K={8..15,24..31})
    v16bf afrag[4];
#pragma unroll
    for (int m = 0; m < 4; ++m) {
      int row = m * 16 + laneN;
      int k0  = ks + (laneHi ? 8 : 0);
      v8bf lo = *(const v8bf*)&sA[row][k0];
      v8bf hi = *(const v8bf*)&sA[row][k0 + 16];
#pragma unroll
      for (int e = 0; e < 8; ++e) { afrag[m][e] = lo[e]; afrag[m][8 + e] = hi[e]; }
    }
#pragma unroll
    for (int h = 0; h < HCNT; ++h) {
      int col = h * DFEAT + dcol;              // 0..1023
      // B fragment (32x16): lane = column; lanes 0-15 K={0..7,16..23}, lanes 16-31 K={8..15,24..31}
      const __bf16* bp = Wt + (((size_t)col) << 8) + ks + (laneHi ? 8 : 0);
      v8bf b0 = *(const v8bf*)bp;
      v8bf b1 = *(const v8bf*)(bp + 16);
      v16bf bfrag;
#pragma unroll
      for (int e = 0; e < 8; ++e) { bfrag[e] = b0[e]; bfrag[8 + e] = b1[e]; }
#pragma unroll
      for (int m = 0; m < 4; ++m)
        acc[m][h] = __builtin_amdgcn_wmma_f32_16x16x32_bf16(
            false, afrag[m], false, bfrag, (short)0, acc[m][h], false, false);
    }
  }

  // a_s / a_d partial dots -> LDS atomics (C layout: VGPR v = row, lane-half = +8)
#pragma unroll
  for (int h = 0; h < HCNT; ++h) {
    float as_w = attS[h * DFEAT + dcol];
    float ad_w = attD[h * DFEAT + dcol];
#pragma unroll
    for (int m = 0; m < 4; ++m) {
#pragma unroll
      for (int v = 0; v < 8; ++v) {
        int row = m * 16 + v + laneHi * 8;
        float hv = acc[m][h][v];
        atomicAdd(&sAS[row][h], hv * as_w);
        atomicAdd(&sAD[row][h], hv * ad_w);
      }
    }
  }
  __syncthreads();

  // per-group softmax over src i for each dst j (graph is K4 + self loops)
  if (tid < 256) {
    int g = tid >> 4, h = (tid >> 2) & 3, j = tid & 3;   // g: 0..15
    float adj = sAD[g * 4 + j][h];
    float e[4], mx = -1e30f;
#pragma unroll
    for (int i = 0; i < 4; ++i) {
      float t = sAS[g * 4 + i][h] + adj;
      e[i] = t > 0.f ? t : NEG_SLOPE * t;
      mx = fmaxf(mx, e[i]);
    }
    float s = 0.f;
#pragma unroll
    for (int i = 0; i < 4; ++i) { e[i] = __expf(e[i] - mx); s += e[i]; }
    float inv = 1.f / s;
#pragma unroll
    for (int i = 0; i < 4; ++i) sAlpha[g][h][j][i] = e[i] * inv;
  }
  __syncthreads();

  // attention mix (pure per-lane register math), head mean, bias, ELU, store
  float bc = bias[dcol];
#pragma unroll
  for (int m = 0; m < 4; ++m) {
    int gA = m * 4 + laneHi * 2;               // group of rows v=0..3 in this half
    float out[8];
#pragma unroll
    for (int v = 0; v < 8; ++v) out[v] = 0.f;
#pragma unroll
    for (int h = 0; h < HCNT; ++h) {
      const v4f* alA = (const v4f*)&sAlpha[gA][h][0][0];
      const v4f* alB = (const v4f*)&sAlpha[gA + 1][h][0][0];
      v8f hc = acc[m][h];
#pragma unroll
      for (int j = 0; j < 4; ++j) {
        v4f a0 = alA[j], a1 = alB[j];
        out[j]     += a0.x * hc[0] + a0.y * hc[1] + a0.z * hc[2] + a0.w * hc[3];
        out[4 + j] += a1.x * hc[4] + a1.y * hc[5] + a1.z * hc[6] + a1.w * hc[7];
      }
    }
#pragma unroll
    for (int v = 0; v < 8; ++v) {
      float val = out[v] * 0.25f + bc;         // head mean (H=4) + bias
      val = val > 0.f ? val : (__expf(val) - 1.f);  // ELU
      size_t row = rowBase + (size_t)(m * 16 + v + laneHi * 8);
      x_out[row * DFEAT + dcol] = val;
    }
  }
}

// ---- readout: fused[b][d] = mean over the 4 nodes ----
__global__ __launch_bounds__(256) void readout_kernel(const float* __restrict__ xn,
                                                      float* __restrict__ fused) {
  size_t id = (size_t)blockIdx.x * 256 + threadIdx.x;   // over B*256
  size_t b  = id >> 8;
  int d     = (int)(id & 255);
  const float* p = xn + (b << 10) + d;
  fused[id] = 0.25f * (p[0] + p[256] + p[512] + p[768]);
}

extern "C" void kernel_launch(void* const* d_in, const int* in_sizes, int n_in,
                              void* d_out, int out_size, void* d_ws, size_t ws_size,
                              hipStream_t stream) {
  const float* spec = (const float*)d_in[0];
  const float* shar = (const float*)d_in[1];
  const float* W    = (const float*)d_in[2];
  const float* attS = (const float*)d_in[3];
  const float* attD = (const float*)d_in[4];
  const float* bias = (const float*)d_in[5];

  const int B = in_sizes[0] / (3 * DFEAT);           // 16384
  const int N = B * 4;

  float* fused = (float*)d_out;                      // [B*256]
  float* xn    = fused + (size_t)B * DFEAT;          // [N*256] (x_nodes output region)
  float* ws_x  = (float*)d_ws;                       // [N*256] ping buffer (64 MB)
  __bf16* Wt   = (__bf16*)((char*)d_ws + (size_t)N * DFEAT * sizeof(float)); // 1 MB

  prep_w_kernel<<<2048, 256, 0, stream>>>(W, Wt);
  pack_kernel<<<(N * 64) / 256, 256, 0, stream>>>(spec, shar, xn, B);
  // layer 0: xn -> ws_x
  gat_layer_kernel<<<N / MT, 512, 0, stream>>>(xn, Wt, attS, attD, bias, ws_x);
  // layer 1: ws_x -> xn (final x_nodes written directly into d_out)
  gat_layer_kernel<<<N / MT, 512, 0, stream>>>(ws_x, Wt + (size_t)1024 * 256,
                                               attS + HCNT * DFEAT, attD + HCNT * DFEAT,
                                               bias + DFEAT, xn);
  readout_kernel<<<B, 256, 0, stream>>>(xn, fused);
}
